// CachedConditionNumberLoss_65343632441792
// MI455X (gfx1250) — matrix operand order repaired and checked by reading, not compiled, not run on hardware
//
#include <hip/hip_runtime.h>
#include <hip/hip_bf16.h>
#include <math.h>

// ---------------------------------------------------------------------------
// CachedConditionNumberLoss for MI455X (gfx1250, wave32, WMMA)
//   A  = F F^T / n + I     (WMMA bf16, f32 accum; A exactly symmetric)
//   L  = I + scatter_add(pred*scale)
//   T  = L A == L A^T      (symmetry -> transposed-B loader, contiguous loads)
//   M  = sym(T L^T)
//   out = log(lam_max(M)) - log(lam_min(M)) via power iteration on M and on
//         (sigma*I - M), sigma ~ 1.02*lam_max; all scalars on-device.
// GEMM: 32x64 block per wave (2x4 WMMA tiles), K-step 32, b128 loads only.
// Power iteration: fused step kernel (normalize + matvec [+shift] + sumsq),
// 2 launches per iteration.
// ---------------------------------------------------------------------------

typedef __bf16 bf16;
typedef __attribute__((ext_vector_type(16))) __bf16 v16bf;
typedef __attribute__((ext_vector_type(2)))  __bf16 bf16x2;
typedef __attribute__((ext_vector_type(8)))  float  v8f;

#define NMAT 2048
#define PI_ITERS 64

// ---------------- conversion ----------------
__global__ void k_f32_to_bf16(const float* __restrict__ src,
                              bf16* __restrict__ dst, int n) {
  int i = blockIdx.x * blockDim.x + threadIdx.x;
  if (i < n) dst[i] = (bf16)src[i];
}

// ---------------- WMMA GEMM: C = alpha * A * Bbuf^T [+ I] ------------------
// A, Bbuf row-major n x n (bf16): C[i][j] = alpha * sum_k A[i][k]*Bbuf[j][k].
// Each wave owns a 32x64 C block: 2 A-frags x 4 B-frags -> 8 WMMAs per
// K-step against 12 contiguous b128 loads. Layouts per CDNA5 ISA 7.12.2.
template <bool ADDI>
__global__ void k_wmma_gemm_bt(const bf16* __restrict__ A,
                               const bf16* __restrict__ B,
                               float* __restrict__ C,
                               int n, float alpha) {
  const int lane = threadIdx.x & 31;
  const int wave = threadIdx.x >> 5;
  const int nblkN = n >> 6;  // 64-wide column blocks
  const int w  = blockIdx.x * (blockDim.x >> 5) + wave;
  const int tm = (w / nblkN) << 5;
  const int tn = (w % nblkN) << 6;
  const int h  = lane >> 4;  // half-wave selects K sub-range
  const int l  = lane & 15;  // M (for A) / N (for B,C)

  v8f acc[2][4];
#pragma unroll
  for (int i = 0; i < 2; ++i)
#pragma unroll
    for (int j = 0; j < 4; ++j) acc[i][j] = (v8f){};

  const bf16* arow[2];
  const bf16* brow[4];
#pragma unroll
  for (int i = 0; i < 2; ++i) arow[i] = A + (size_t)(tm + 16 * i + l) * n;
#pragma unroll
  for (int j = 0; j < 4; ++j) brow[j] = B + (size_t)(tn + 16 * j + l) * n;

  for (int k0 = 0; k0 < n; k0 += 32) {
    // prefetch two K-steps ahead (speculative; OOB prefetch silently dropped)
#pragma unroll
    for (int i = 0; i < 2; ++i) __builtin_prefetch(arow[i] + k0 + 64, 0, 1);
#pragma unroll
    for (int j = 0; j < 4; ++j) __builtin_prefetch(brow[j] + k0 + 64, 0, 1);

    v16bf af[2], bfr[4];
    // A fragment: lanes 0-15 K {0..7,16..23}, lanes 16-31 K {8..15,24..31}
#pragma unroll
    for (int i = 0; i < 2; ++i) {
#pragma unroll
      for (int v = 0; v < 8; ++v) {
        const int kk = k0 + ((v < 4) ? (2 * v + 8 * h) : (16 + 2 * (v - 4) + 8 * h));
        bf16x2 p = *(const bf16x2*)(arow[i] + kk);
        af[i][2 * v] = p.x; af[i][2 * v + 1] = p.y;
      }
    }
    // B fragment (B^T read): lanes 0-15 K {0..7,16..23}, lanes 16-31 K {8..15,24..31}
#pragma unroll
    for (int j = 0; j < 4; ++j) {
#pragma unroll
      for (int v = 0; v < 8; ++v) {
        const int kk = k0 + ((v < 4) ? (2 * v + 8 * h) : (16 + 2 * (v - 4) + 8 * h));
        bf16x2 p = *(const bf16x2*)(brow[j] + kk);
        bfr[j][2 * v] = p.x; bfr[j][2 * v + 1] = p.y;
      }
    }
#pragma unroll
    for (int i = 0; i < 2; ++i)
#pragma unroll
      for (int j = 0; j < 4; ++j)
        acc[i][j] = __builtin_amdgcn_wmma_f32_16x16x32_bf16(
            false, af[i], false, bfr[j], (short)0, acc[i][j], false, false);
  }

#pragma unroll
  for (int v = 0; v < 8; ++v) {
#pragma unroll
    for (int i = 0; i < 2; ++i) {
      const int row = tm + 16 * i + v + 8 * h;  // C/D: VGPR v -> M = v + 8*half
#pragma unroll
      for (int j = 0; j < 4; ++j) {
        const int col = tn + 16 * j + l;
        float d = alpha * acc[i][j][v];
        if (ADDI && row == col) d += 1.0f;
        C[(size_t)row * n + col] = d;
      }
    }
  }
}

// ---------------- L assembly ----------------
__global__ void k_set_identity(float* __restrict__ L, int n) {
  int i = blockIdx.x * blockDim.x + threadIdx.x;
  if (i < n * n) L[i] = ((i / n) == (i % n)) ? 1.0f : 0.0f;
}

__global__ void k_scatter_add(float* __restrict__ L,
                              const float* __restrict__ pv,
                              const float* __restrict__ sc,
                              const int* __restrict__ r,
                              const int* __restrict__ c, int nnz, int n) {
  int i = blockIdx.x * blockDim.x + threadIdx.x;
  if (i < nnz) atomicAdd(&L[(size_t)r[i] * n + c[i]], pv[i] * sc[i]);
}

__global__ void k_symmetrize(float* __restrict__ M, int n) {
  int i = blockIdx.x * blockDim.x + threadIdx.x;
  if (i >= n * n) return;
  int rr = i / n, cc = i % n;
  if (rr < cc) {
    float a = M[i], b = M[(size_t)cc * n + rr];
    float m = 0.5f * (a + b);
    M[i] = m;
    M[(size_t)cc * n + rr] = m;
  }
}

// ---------------- fused power-iteration step ----------------
// Previous launch finalized s[2] = 1/||x|| (x un-normalized). This kernel
// computes, per row:  t = (M x)[row]
//   SHIFTED=0: y[row] = s2 * t                    (power iter on M)
//   SHIFTED=1: y[row] = s2 * (s4 * x[row] - t)    (power iter on sigma*I - M)
// and accumulates y[row]^2 into s[0] (zeroed by the previous finalize).
template <bool SHIFTED>
__global__ void k_pi_step(const float* __restrict__ M, const float* __restrict__ x,
                          float* __restrict__ y, float* __restrict__ s, int n) {
  __shared__ float sm[256];
  const int row = blockIdx.x;
  const float4* mr = (const float4*)(M + (size_t)row * n);
  const float4* x4 = (const float4*)x;
  float acc = 0.f;
  for (int j = threadIdx.x; j < (n >> 2); j += blockDim.x) {
    float4 m = mr[j], xx = x4[j];
    acc += m.x * xx.x + m.y * xx.y + m.z * xx.z + m.w * xx.w;
  }
  sm[threadIdx.x] = acc;
  __syncthreads();
  for (int st = 128; st > 0; st >>= 1) {
    if (threadIdx.x < st) sm[threadIdx.x] += sm[threadIdx.x + st];
    __syncthreads();
  }
  if (threadIdx.x == 0) {
    const float s2 = s[2];
    float t = sm[0];
    float z = SHIFTED ? s2 * (s[4] * x[row] - t) : s2 * t;
    y[row] = z;
    atomicAdd(s, z * z);
  }
}

// s[1] = ||y|| (eigenvalue estimate: input was normalized), s[2] = 1/||y||,
// and reset the accumulator for the next step.
__global__ void k_finalize_norm(float* s) {
  float lam = sqrtf(fmaxf(s[0], 1e-30f));
  s[1] = lam;
  s[2] = 1.f / lam;
  s[0] = 0.f;
}

// init start vector; reset accumulator and set unit scale for first step
__global__ void k_fill(float* __restrict__ v, float* __restrict__ s, int n) {
  int i = blockIdx.x * blockDim.x + threadIdx.x;
  if (i < n) v[i] = 1.0f + 1e-4f * (float)(i & 31);
  if (i == 0) { s[0] = 0.f; s[2] = 1.f; }
}

__global__ void k_after_max(float* s) {
  s[3] = s[1];                  // lam_max
  s[4] = s[1] * 1.02f + 1e-6f;  // shift sigma > lam_max
}
__global__ void k_after_min(float* s) { s[5] = s[1]; }  // mu = sigma - lam_min

__global__ void k_final(float* __restrict__ out, const float* __restrict__ s) {
  float lmax = fmaxf(s[3], 1e-12f);
  float lmin = fmaxf(s[4] - s[5], 1e-12f);
  out[0] = logf(lmax) - logf(lmin);
}

// ---------------------------------------------------------------------------
extern "C" void kernel_launch(void* const* d_in, const int* in_sizes, int n_in,
                              void* d_out, int out_size, void* d_ws, size_t ws_size,
                              hipStream_t stream) {
  (void)n_in; (void)out_size; (void)ws_size;
  const int n = NMAT;
  const int nnz = in_sizes[0];
  const float* pred   = (const float*)d_in[0];
  const float* scales = (const float*)d_in[1];
  const float* Afac   = (const float*)d_in[2];
  const int*   rows   = (const int*)d_in[3];
  const int*   cols   = (const int*)d_in[4];

  const size_t elems = (size_t)n * n;
  char* w = (char*)d_ws;
  bf16*  Fb   = (bf16*)(w);                           // F (bf16); later Tb
  bf16*  Ab   = (bf16*)(w + elems * 2);               // A (bf16, symmetric)
  bf16*  Lb   = (bf16*)(w + elems * 4);               // L (bf16)
  float* Amat = (float*)(w + elems * 6);              // A (f32); later T = L*A
  float* Lmat = (float*)(w + elems * 6 + elems * 4);  // L (f32); later M
  float* x0   = (float*)(w + elems * 6 + elems * 8);  // ping-pong iterates
  float* x1   = x0 + n;
  float* scal = x1 + n;  // [0]=sumsq [1]=lam [2]=1/lam [3]=lam_max [4]=shift [5]=mu
  bf16*  Tb   = Fb;
  float* Tmat = Amat;
  float* Mmat = Lmat;

  const int cvtBlocks  = (int)((elems + 255) / 256);
  const int gemmBlocks = (n / 32) * (n / 64) / 8;  // 8 waves (32x64 blocks)/block
  const int vb = (n + 255) / 256;

  // A = F F^T / n + I   (exactly symmetric by construction)
  k_f32_to_bf16<<<cvtBlocks, 256, 0, stream>>>(Afac, Fb, (int)elems);
  k_wmma_gemm_bt<true><<<gemmBlocks, 256, 0, stream>>>(Fb, Fb, Amat, n, 1.0f / (float)n);

  // L = I + scatter
  k_set_identity<<<cvtBlocks, 256, 0, stream>>>(Lmat, n);
  k_scatter_add<<<(nnz + 255) / 256, 256, 0, stream>>>(Lmat, pred, scales, rows, cols, nnz, n);

  // T = L * A = L * A^T  (A symmetric -> contiguous BT loads)
  k_f32_to_bf16<<<cvtBlocks, 256, 0, stream>>>(Amat, Ab, (int)elems);
  k_f32_to_bf16<<<cvtBlocks, 256, 0, stream>>>(Lmat, Lb, (int)elems);
  k_wmma_gemm_bt<false><<<gemmBlocks, 256, 0, stream>>>(Lb, Ab, Tmat, n, 1.0f);
  // M = T * L^T
  k_f32_to_bf16<<<cvtBlocks, 256, 0, stream>>>(Tmat, Tb, (int)elems);
  k_wmma_gemm_bt<false><<<gemmBlocks, 256, 0, stream>>>(Tb, Lb, Mmat, n, 1.0f);
  k_symmetrize<<<cvtBlocks, 256, 0, stream>>>(Mmat, n);

  // lam_max via power iteration (2 launches/iter)
  k_fill<<<vb, 256, 0, stream>>>(x0, scal, n);
  {
    float* xin = x0; float* xout = x1;
    for (int it = 0; it < PI_ITERS; ++it) {
      k_pi_step<false><<<n, 256, 0, stream>>>(Mmat, xin, xout, scal, n);
      k_finalize_norm<<<1, 1, 0, stream>>>(scal);
      float* tmp = xin; xin = xout; xout = tmp;
    }
  }
  k_after_max<<<1, 1, 0, stream>>>(scal);

  // lam_min via power iteration on (shift*I - M) (2 launches/iter)
  k_fill<<<vb, 256, 0, stream>>>(x0, scal, n);
  {
    float* xin = x0; float* xout = x1;
    for (int it = 0; it < PI_ITERS; ++it) {
      k_pi_step<true><<<n, 256, 0, stream>>>(Mmat, xin, xout, scal, n);
      k_finalize_norm<<<1, 1, 0, stream>>>(scal);
      float* tmp = xin; xin = xout; xout = tmp;
    }
  }
  k_after_min<<<1, 1, 0, stream>>>(scal);

  k_final<<<1, 1, 0, stream>>>((float*)d_out, scal);
}
